// SparseCINCochainConv_56238301774323
// MI455X (gfx1250) — matrix operand order repaired and checked by reading, not compiled, not run on hardware
//
#include <hip/hip_runtime.h>
#include <hip/hip_bf16.h>
#include <hip/hip_fp16.h>

// ---------------------------------------------------------------------------
// MI455X (gfx1250) implementation of SparseCINCochainConv forward.
// wave32, WMMA f32_16x16x32_f16. Design rationale:
//  * x/boundary_attr (51MB) are L2-resident (192MB L2) -> edge gathers are L2
//    traffic, not HBM. So we RECOMPUTE the edge GEMM (pass1: BN stats only,
//    pass2: BN+ReLU apply + atomic scatter from registers) instead of spilling
//    a 205MB pre-BN message buffer to HBM. Saves ~820MB HBM round-trip.
//  * f16 WMMA with fp32 accumulation: 57 GFLOP total is negligible vs memory.
//  * Linear bias cancels exactly under training-mode BatchNorm -> skipped.
//  * BN(prev)+ReLU is fused into the A-operand load of the next GEMM.
// ---------------------------------------------------------------------------

typedef __attribute__((ext_vector_type(16))) _Float16 v16h;
typedef __attribute__((ext_vector_type(8)))  float    v8f;

union Frag16 { v16h v; unsigned u[8]; _Float16 h[16]; };

__device__ __forceinline__ unsigned pack2h(float a, float b) {
  union { unsigned u; _Float16 h[2]; } p;
  p.h[0] = (_Float16)a; p.h[1] = (_Float16)b;
  return p.u;
}

#define BN_EPS 1e-5f

// Transpose-convert weight [K][64] f32 -> Wt [64][K] f16
__global__ void convert_w(const float* __restrict__ W, _Float16* __restrict__ Wt, int K) {
  int t = blockIdx.x * blockDim.x + threadIdx.x;
  if (t >= K * 64) return;
  int c = t / K, k = t % K;
  Wt[c * K + k] = (_Float16)W[(size_t)k * 64 + c];
}

// ---------------------------------------------------------------------------
// Edge message GEMM: h[e] = cat(featA[idxA[e]], featB[idxB[e]]) @ W  (128->64)
// 128 threads = 4 waves; wave w computes output channels [16w,16w+16).
// Each block handles 4 tiles of 16 edges.
//   APPLY=false: accumulate per-channel BN sum/sumsq only (nothing stored).
//   APPLY=true : out[tgt[e]] += ReLU(h[e]*scale+shift), straight from the
//                WMMA accumulator fragments (L2 float atomics).
// ---------------------------------------------------------------------------
template <bool APPLY>
__global__ __launch_bounds__(128)
void edge_msg_gemm(const float* __restrict__ featA, const int* __restrict__ idxA,
                   const float* __restrict__ featB, const int* __restrict__ idxB,
                   const _Float16* __restrict__ Wt,   // [64][128] f16
                   float* __restrict__ stats,         // [128]: sum, sumsq (!APPLY)
                   const float* __restrict__ ss,      // [128]: scale, shift (APPLY)
                   const int* __restrict__ tgt,       // scatter rows (APPLY)
                   float* __restrict__ out,           // [N][64] f32 (APPLY)
                   int nTiles)
{
  __shared__ _Float16 lA[16][136];   // 16 edges x 128 feats, +8 pad (bank skew)
  __shared__ _Float16 lW[64][136];   // 64 out-ch x 128 K, transposed, +8 pad

  const int tid  = threadIdx.x;
  const int lane = tid & 31;
  const int wave = tid >> 5;          // N-tile
  const int g    = lane >> 4;         // half-wave group
  const int mrow = lane & 15;         // A row this lane loads
  const int ch   = (lane & 15) + 16 * wave;  // output channel (B/D column)

  // Stage transposed weights into LDS (uint4 = 8 halfs per copy)
  for (int i = tid; i < 64 * 16; i += 128) {
    int r = i >> 4, c8 = i & 15;
    *(uint4*)(&lW[r][c8 * 8]) = *(const uint4*)(Wt + r * 128 + c8 * 8);
  }

  float ssum = 0.f, ssq = 0.f;
  float sc = 0.f, sh = 0.f;
  if (APPLY) { sc = ss[ch]; sh = ss[64 + ch]; }

  const int tile0 = blockIdx.x * 4;

  for (int it = 0; it < 4; ++it) {
    int tile = tile0 + it;
    if (tile >= nTiles) break;           // uniform per block
    int ebase = tile * 16;

    // Gather: 8 threads per edge, 16 feats each (4x float4), convert to f16
    {
      int et = tid >> 3;
      int fb = (tid & 7) * 16;
      int e  = ebase + et;
      const float* src;
      if (fb < 64) src = featA + (size_t)idxA[e] * 64 + fb;
      else         src = featB + (size_t)idxB[e] * 64 + (fb - 64);
      float4 v0 = ((const float4*)src)[0];
      float4 v1 = ((const float4*)src)[1];
      float4 v2 = ((const float4*)src)[2];
      float4 v3 = ((const float4*)src)[3];
      unsigned* dst = (unsigned*)(&lA[et][fb]);
      dst[0] = pack2h(v0.x, v0.y); dst[1] = pack2h(v0.z, v0.w);
      dst[2] = pack2h(v1.x, v1.y); dst[3] = pack2h(v1.z, v1.w);
      dst[4] = pack2h(v2.x, v2.y); dst[5] = pack2h(v2.z, v2.w);
      dst[6] = pack2h(v3.x, v3.y); dst[7] = pack2h(v3.z, v3.w);
    }
    __syncthreads();

    v8f acc = {0.f,0.f,0.f,0.f,0.f,0.f,0.f,0.f};
    #pragma unroll
    for (int k0 = 0; k0 < 128; k0 += 32) {
      Frag16 a, b;
      // A 16x32 f16 layout: lane m, element j -> K = j + 16g (contiguous)
      const unsigned* pa = (const unsigned*)(&lA[mrow][k0]);
      #pragma unroll
      for (int t2 = 0; t2 < 4; ++t2) {
        a.u[t2]     = pa[t2 + 8 * g];
        a.u[4 + t2] = pa[4 + t2 + 8 * g];
      }
      // B 32x16 f16 layout: lane n=ch, element j -> K = j + 16g (contiguous)
      const unsigned* pb = (const unsigned*)(&lW[ch][k0 + 16 * g]);
      #pragma unroll
      for (int t2 = 0; t2 < 8; ++t2) b.u[t2] = pb[t2];
      acc = __builtin_amdgcn_wmma_f32_16x16x32_f16(false, a.v, false, b.v,
                                                   (short)0, acc, false, false);
    }
    // D layout: vgpr r -> row m = r + 8g, column = ch
    if (APPLY) {
      #pragma unroll
      for (int r = 0; r < 8; ++r) {
        int e = ebase + r + 8 * g;
        float v = fmaxf(0.f, acc[r] * sc + sh);
        atomicAdd(&out[(size_t)tgt[e] * 64 + ch], v);
      }
    } else {
      #pragma unroll
      for (int r = 0; r < 8; ++r) {
        float d = acc[r];
        ssum += d; ssq += d * d;
      }
    }
    __syncthreads();
  }

  if (!APPLY) {
    // Per-channel BN partial stats: combine half-waves, then 128 atomics/block
    ssum += __shfl_xor(ssum, 16, 32);
    ssq  += __shfl_xor(ssq, 16, 32);
    if (lane < 16) {
      atomicAdd(&stats[ch], ssum);
      atomicAdd(&stats[64 + ch], ssq);
    }
  }
}

// ---------------------------------------------------------------------------
// Dense row GEMM [M,K] @ [K,64] with optional fused BN+ReLU on the A sources.
// K=64 (update blocks) or K=128 (combine: A0=up half, A1=boundary half).
// ---------------------------------------------------------------------------
template <int K>
__global__ __launch_bounds__(128)
void row_gemm_bn(const float* __restrict__ A0, const float* __restrict__ ss0,
                 const float* __restrict__ A1, const float* __restrict__ ss1,
                 const _Float16* __restrict__ Wt,   // [64][K] f16
                 float* __restrict__ out,           // [M][64] f32 pre-BN
                 float* __restrict__ stats,         // [128]
                 int nTiles)
{
  __shared__ _Float16 lA[16][K + 8];
  __shared__ _Float16 lW[64][K + 8];

  const int tid  = threadIdx.x;
  const int lane = tid & 31;
  const int wave = tid >> 5;
  const int g    = lane >> 4;
  const int mrow = lane & 15;
  const int ch   = (lane & 15) + 16 * wave;

  constexpr int W8 = K / 8;
  for (int i = tid; i < 64 * W8; i += 128) {
    int r = i / W8, c8 = i % W8;
    *(uint4*)(&lW[r][c8 * 8]) = *(const uint4*)(Wt + r * K + c8 * 8);
  }

  float ssum = 0.f, ssq = 0.f;
  const int tile0 = blockIdx.x * 4;
  constexpr int CPR = K / 4;   // float4 chunks per row

  for (int it = 0; it < 4; ++it) {
    int tile = tile0 + it;
    if (tile >= nTiles) break;
    int ebase = tile * 16;

    for (int i = tid; i < 16 * CPR; i += 128) {
      int rt = i / CPR;
      int f  = (i % CPR) * 4;
      size_t row = (size_t)(ebase + rt);
      float4 v; const float* ssp; int cb;
      if (f < 64) { v = *(const float4*)(A0 + row * 64 + f);        ssp = ss0; cb = f; }
      else        { v = *(const float4*)(A1 + row * 64 + (f - 64)); ssp = ss1; cb = f - 64; }
      if (ssp) {  // fused BN + ReLU of the previous block
        v.x = fmaxf(0.f, v.x * ssp[cb + 0] + ssp[64 + cb + 0]);
        v.y = fmaxf(0.f, v.y * ssp[cb + 1] + ssp[64 + cb + 1]);
        v.z = fmaxf(0.f, v.z * ssp[cb + 2] + ssp[64 + cb + 2]);
        v.w = fmaxf(0.f, v.w * ssp[cb + 3] + ssp[64 + cb + 3]);
      }
      unsigned* dst = (unsigned*)(&lA[rt][f]);
      dst[0] = pack2h(v.x, v.y);
      dst[1] = pack2h(v.z, v.w);
    }
    __syncthreads();

    v8f acc = {0.f,0.f,0.f,0.f,0.f,0.f,0.f,0.f};
    #pragma unroll
    for (int k0 = 0; k0 < K; k0 += 32) {
      Frag16 a, b;
      const unsigned* pa = (const unsigned*)(&lA[mrow][k0]);
      #pragma unroll
      for (int t2 = 0; t2 < 4; ++t2) {
        a.u[t2]     = pa[t2 + 8 * g];
        a.u[4 + t2] = pa[4 + t2 + 8 * g];
      }
      const unsigned* pb = (const unsigned*)(&lW[ch][k0 + 16 * g]);
      #pragma unroll
      for (int t2 = 0; t2 < 8; ++t2) b.u[t2] = pb[t2];
      acc = __builtin_amdgcn_wmma_f32_16x16x32_f16(false, a.v, false, b.v,
                                                   (short)0, acc, false, false);
    }
    #pragma unroll
    for (int r = 0; r < 8; ++r) {
      float d = acc[r];
      ssum += d; ssq += d * d;
      int m = r + 8 * g;
      out[(size_t)(ebase + m) * 64 + ch] = d;
    }
    __syncthreads();
  }

  ssum += __shfl_xor(ssum, 16, 32);
  ssq  += __shfl_xor(ssq, 16, 32);
  if (lane < 16) {
    atomicAdd(&stats[ch], ssum);
    atomicAdd(&stats[64 + ch], ssq);
  }
}

// stats -> (scale, shift):  scale = g*rsqrt(var+eps); shift = be - mean*scale
__global__ void bn_finalize(const float* __restrict__ st,
                            const float* __restrict__ gg,
                            const float* __restrict__ be,
                            float invCnt, float* __restrict__ ss) {
  int c = threadIdx.x;   // 64 threads
  float mean = st[c] * invCnt;
  float var  = st[64 + c] * invCnt - mean * mean;
  float sc   = gg[c] * rsqrtf(var + BN_EPS);
  ss[c]      = sc;
  ss[64 + c] = be[c] - mean * sc;
}

// out = (1 + eps) * x
__global__ void init_out(float* __restrict__ out, const float* __restrict__ x,
                         const float* __restrict__ eps, int n) {
  int i = blockIdx.x * blockDim.x + threadIdx.x;
  if (i >= n) return;
  out[i] = (1.0f + eps[0]) * x[i];
}

// d_out = ReLU(h*scale + shift)
__global__ void final_apply(const float* __restrict__ h, const float* __restrict__ ss,
                            float* __restrict__ out, int n) {
  int i = blockIdx.x * blockDim.x + threadIdx.x;
  if (i >= n) return;
  int c = i & 63;
  out[i] = fmaxf(0.f, h[i] * ss[c] + ss[64 + c]);
}

// ---------------------------------------------------------------------------
extern "C" void kernel_launch(void* const* d_in, const int* in_sizes, int n_in,
                              void* d_out, int out_size, void* d_ws, size_t ws_size,
                              hipStream_t stream) {
  (void)n_in; (void)out_size; (void)ws_size;
  // Inputs flattened in setup_inputs() insertion order (nested dicts DFS):
  const float* x     = (const float*)d_in[0];
  const float* battr = (const float*)d_in[1];
  const float* W_mu  = (const float*)d_in[2];   // msg_up.lin.W  [128,64]
  const float* g_mu  = (const float*)d_in[4];
  const float* be_mu = (const float*)d_in[5];
  const float* W_mb  = (const float*)d_in[6];   // msg_b.lin.W
  const float* g_mb  = (const float*)d_in[8];
  const float* be_mb = (const float*)d_in[9];
  const float* W_u0  = (const float*)d_in[10];  // upd_up[0]
  const float* g_u0  = (const float*)d_in[12];
  const float* be_u0 = (const float*)d_in[13];
  const float* W_u1  = (const float*)d_in[14];  // upd_up[1]
  const float* g_u1  = (const float*)d_in[16];
  const float* be_u1 = (const float*)d_in[17];
  const float* W_b0  = (const float*)d_in[18];  // upd_b[0]
  const float* g_b0  = (const float*)d_in[20];
  const float* be_b0 = (const float*)d_in[21];
  const float* W_b1  = (const float*)d_in[22];  // upd_b[1]
  const float* g_b1  = (const float*)d_in[24];
  const float* be_b1 = (const float*)d_in[25];
  const float* W_cb  = (const float*)d_in[26];  // combine [128,64]
  const float* g_cb  = (const float*)d_in[28];
  const float* be_cb = (const float*)d_in[29];
  const float* eps1  = (const float*)d_in[30];
  const float* eps2  = (const float*)d_in[31];
  const int*   upI   = (const int*)d_in[32];    // [2,E]
  const int*   bdI   = (const int*)d_in[33];    // [2,E]

  const int N = in_sizes[0] / 64;
  const int E = in_sizes[32] / 2;

  char* ws = (char*)d_ws;
  _Float16* WtMU = (_Float16*)(ws + 0);
  _Float16* WtMB = (_Float16*)(ws + 16384);
  _Float16* WtU0 = (_Float16*)(ws + 32768);
  _Float16* WtU1 = (_Float16*)(ws + 40960);
  _Float16* WtB0 = (_Float16*)(ws + 49152);
  _Float16* WtB1 = (_Float16*)(ws + 57344);
  _Float16* WtCB = (_Float16*)(ws + 65536);
  float* stats = (float*)(ws + 81920);   // 7 x 128 floats
  float* ssb   = (float*)(ws + 85504);   // 7 x 128 floats
  size_t off = 89088;
  float* bufA = (float*)(ws + off); off += (size_t)N * 64 * 4;  // out_up / out_b
  float* htmp = (float*)(ws + off); off += (size_t)N * 64 * 4;  // h1 / h3
  float* h2u  = (float*)(ws + off); off += (size_t)N * 64 * 4;
  float* h2b  = (float*)(ws + off); off += (size_t)N * 64 * 4;

  hipMemsetAsync(stats, 0, 7 * 128 * sizeof(float), stream);

  convert_w<<<dim3((128 * 64 + 255) / 256), 256, 0, stream>>>(W_mu, WtMU, 128);
  convert_w<<<dim3((128 * 64 + 255) / 256), 256, 0, stream>>>(W_mb, WtMB, 128);
  convert_w<<<dim3((64 * 64 + 255) / 256), 256, 0, stream>>>(W_u0, WtU0, 64);
  convert_w<<<dim3((64 * 64 + 255) / 256), 256, 0, stream>>>(W_u1, WtU1, 64);
  convert_w<<<dim3((64 * 64 + 255) / 256), 256, 0, stream>>>(W_b0, WtB0, 64);
  convert_w<<<dim3((64 * 64 + 255) / 256), 256, 0, stream>>>(W_b1, WtB1, 64);
  convert_w<<<dim3((128 * 64 + 255) / 256), 256, 0, stream>>>(W_cb, WtCB, 128);

  const int nTE = E / 16, gE = (nTE + 3) / 4;
  const int nTN = N / 16, gD = (nTN + 3) / 4;
  const int gElem = (N * 64 + 255) / 256;
  const float invE = 1.0f / (float)E, invN = 1.0f / (float)N;

  // ---- UP path: cat(x[up0], x[up1]) -> BN stats -> recompute+scatter on up0
  edge_msg_gemm<false><<<gE, 128, 0, stream>>>(x, upI, x, upI + E, WtMU,
                                               stats + 0, nullptr, nullptr, nullptr, nTE);
  bn_finalize<<<1, 64, 0, stream>>>(stats + 0, g_mu, be_mu, invE, ssb + 0);
  init_out<<<gElem, 256, 0, stream>>>(bufA, x, eps1, N * 64);
  edge_msg_gemm<true><<<gE, 128, 0, stream>>>(x, upI, x, upI + E, WtMU,
                                              nullptr, ssb + 0, upI, bufA, nTE);
  row_gemm_bn<64><<<gD, 128, 0, stream>>>(bufA, nullptr, nullptr, nullptr, WtU0, htmp, stats + 128, nTN);
  bn_finalize<<<1, 64, 0, stream>>>(stats + 128, g_u0, be_u0, invN, ssb + 128);
  row_gemm_bn<64><<<gD, 128, 0, stream>>>(htmp, ssb + 128, nullptr, nullptr, WtU1, h2u, stats + 256, nTN);
  bn_finalize<<<1, 64, 0, stream>>>(stats + 256, g_u1, be_u1, invN, ssb + 256);  // applied in combine

  // ---- BOUNDARY path: cat(x[b1], battr[b0]) -> stats -> recompute+scatter on b1
  edge_msg_gemm<false><<<gE, 128, 0, stream>>>(x, bdI + E, battr, bdI, WtMB,
                                               stats + 384, nullptr, nullptr, nullptr, nTE);
  bn_finalize<<<1, 64, 0, stream>>>(stats + 384, g_mb, be_mb, invE, ssb + 384);
  init_out<<<gElem, 256, 0, stream>>>(bufA, x, eps2, N * 64);
  edge_msg_gemm<true><<<gE, 128, 0, stream>>>(x, bdI + E, battr, bdI, WtMB,
                                              nullptr, ssb + 384, bdI + E, bufA, nTE);
  row_gemm_bn<64><<<gD, 128, 0, stream>>>(bufA, nullptr, nullptr, nullptr, WtB0, htmp, stats + 512, nTN);
  bn_finalize<<<1, 64, 0, stream>>>(stats + 512, g_b0, be_b0, invN, ssb + 512);
  row_gemm_bn<64><<<gD, 128, 0, stream>>>(htmp, ssb + 512, nullptr, nullptr, WtB1, h2b, stats + 640, nTN);
  bn_finalize<<<1, 64, 0, stream>>>(stats + 640, g_b1, be_b1, invN, ssb + 640);

  // ---- combine: BN+ReLU fused on both halves of the concat ----
  row_gemm_bn<128><<<gD, 128, 0, stream>>>(h2u, ssb + 256, h2b, ssb + 640, WtCB, htmp, stats + 768, nTN);
  bn_finalize<<<1, 64, 0, stream>>>(stats + 768, g_cb, be_cb, invN, ssb + 768);
  final_apply<<<gElem, 256, 0, stream>>>(htmp, ssb + 768, (float*)d_out, N * 64);
}
